// ConditionalNSF_8151847928059
// MI455X (gfx1250) — matrix operand order repaired and checked by reading, not compiled, not run on hardware
//
#include <hip/hip_runtime.h>
#include <hip/hip_bf16.h>
#include <math.h>

// ---------------------------------------------------------------------------
// Conditional NSF coupling flow, fused single kernel for gfx1250 (MI455X).
// - 16-row tiles per wave32; all MLP layers via v_wmma_f32_16x16x32_f16
//   (f16 A/B, f32 accumulate), 9 WMMAs per MLP, 18 per tile.
// - x tile staged LDS-direct with global_load_async_to_lds_b128, cond with
//   _b32 (ASYNCcnt path), overlapped with LDS zero-pad init.
// - A-fragments from zero-padded LDS tiles (unconditional ds loads);
//   B-fragments via clamped addresses + value masks; hidden-state stores
//   value-masked (zero exec divergence in the whole MLP).
// - RQS spline math fully branch-free fp32 VALU (v_exp/v_log/v_rcp).
// - logdet reduced per row with ds_add_f32 atomics; outputs written b128.
// ---------------------------------------------------------------------------

typedef __attribute__((ext_vector_type(16))) _Float16 v16h;
typedef __attribute__((ext_vector_type(8)))  float    v8f;

#define KBINS   5
#define TAILB   3.0f
#define MINBW   0.001f
#define MINBH   0.001f
#define MIND    0.001f
#define WAVES_PER_BLOCK 4
#define ROWS_PER_WAVE   16

__device__ __forceinline__ float frcp(float v) { return __builtin_amdgcn_rcpf(v); }

__device__ __forceinline__ float softplus_f(float v) {
    float e = __expf(fminf(v, 30.0f));
    float r = __logf(1.0f + e);
    return (v > 30.0f) ? v : r;           // v_cndmask, branch-free
}

__device__ __forceinline__ float tanh_f(float v) {
    float xc = fminf(fmaxf(v, -15.0f), 15.0f);
    float e  = __expf(2.0f * xc);
    return (e - 1.0f) * frcp(e + 1.0f);   // branch-free
}

// LDS byte offset of a __shared__ object (generic addr low 32 bits).
template <typename T>
__device__ __forceinline__ uint32_t lds_off(T* p) {
    return (uint32_t)(uintptr_t)p;
}

// Async LDS-direct global loads (GVS mode): LDS[ldsaddr] = MEM[base + voff].
__device__ __forceinline__ void async_ld_b32(uint32_t ldsaddr, uint32_t voff,
                                             const void* base) {
    asm volatile("global_load_async_to_lds_b32 %0, %1, %2"
                 :: "v"(ldsaddr), "v"(voff), "s"(base) : "memory");
}
__device__ __forceinline__ void async_ld_b128(uint32_t ldsaddr, uint32_t voff,
                                              const void* base) {
    asm volatile("global_load_async_to_lds_b128 %0, %1, %2"
                 :: "v"(ldsaddr), "v"(voff), "s"(base) : "memory");
}
__device__ __forceinline__ void wait_async0() {
    asm volatile("s_wait_asynccnt 0" ::: "memory");
}

// K index for pair p (0..7) of the 16-bit A fragment (16x32, MxK layout)
__device__ __forceinline__ int a_kmap(int p, int khalf) {
    int k = (p < 4) ? (2 * p) : (16 + 2 * (p - 4));
    return k + 8 * khalf;
}

__device__ __forceinline__ void softmax5(const float* in, float scale, float* outp) {
    float m = in[0];
#pragma unroll
    for (int i = 1; i < 5; ++i) m = fmaxf(m, in[i]);
    float e[5]; float s = 0.0f;
#pragma unroll
    for (int i = 0; i < 5; ++i) { e[i] = __expf(in[i] - m); s += e[i]; }
    float inv = scale * frcp(s);
#pragma unroll
    for (int i = 0; i < 5; ++i) outp[i] = e[i] * inv;
}

// Rational-quadratic spline with linear tails, faithful to the reference
// (double softmax on W/H, nested softplus on derivatives). Branch-free.
__device__ __forceinline__ void rqs_eval(const float* prm, float xval,
                                         float& yout, float& lad) {
    float Wp[5], Hp[5], wd[5], ht[5];
    softmax5(prm,     2.0f * TAILB, Wp);  // 6*softmax (spline_params stage)
    softmax5(prm + 5, 2.0f * TAILB, Hp);
    softmax5(Wp, 1.0f, wd);               // inner softmax (rqs stage)
    softmax5(Hp, 1.0f, ht);

    float cw[6], ch[6];
    cw[0] = -TAILB; ch[0] = -TAILB;
    {
        float acc = 0.0f;
#pragma unroll
        for (int i = 0; i < 5; ++i) {
            acc += MINBW + (1.0f - MINBW * (float)KBINS) * wd[i];
            cw[i + 1] = 2.0f * TAILB * acc - TAILB;
        }
        cw[5] = TAILB;
#pragma unroll
        for (int i = 0; i < 5; ++i) wd[i] = cw[i + 1] - cw[i];
    }
    {
        float acc = 0.0f;
#pragma unroll
        for (int i = 0; i < 5; ++i) {
            acc += MINBH + (1.0f - MINBH * (float)KBINS) * ht[i];
            ch[i + 1] = 2.0f * TAILB * acc - TAILB;
        }
        ch[5] = TAILB;
#pragma unroll
        for (int i = 0; i < 5; ++i) ht[i] = ch[i + 1] - ch[i];
    }

    float dv[6];
    dv[0] = 1.0f; dv[5] = 1.0f;  // boundary pad: MIN_D + softplus(const) == 1
#pragma unroll
    for (int i = 0; i < 4; ++i) dv[i + 1] = MIND + softplus_f(softplus_f(prm[10 + i]));

    bool inside = (xval >= -TAILB) && (xval <= TAILB);
    float xc = fminf(fmaxf(xval, -TAILB), TAILB);

    int idx = -1;
#pragma unroll
    for (int i = 0; i < 6; ++i) idx += (xc >= cw[i]) ? 1 : 0;
    idx = min(max(idx, 0), KBINS - 1);

    // gather via compare-select chain (stays in VGPRs)
    float in_cw = cw[0], in_w = wd[0], in_ch = ch[0], in_h = ht[0];
    float in_d = dv[0], in_dp1 = dv[1];
#pragma unroll
    for (int i = 1; i < 5; ++i) {
        bool s = (idx == i);
        in_cw  = s ? cw[i] : in_cw;  in_w   = s ? wd[i] : in_w;
        in_ch  = s ? ch[i] : in_ch;  in_h   = s ? ht[i] : in_h;
        in_d   = s ? dv[i] : in_d;   in_dp1 = s ? dv[i + 1] : in_dp1;
    }
    float rw = frcp(in_w);
    float in_delta = in_h * rw;
    float theta = (xc - in_cw) * rw;
    float omt   = 1.0f - theta;
    float tomt  = theta * omt;
    float num   = in_h * (in_delta * theta * theta + in_d * tomt);
    float den   = in_delta + (in_d + in_dp1 - 2.0f * in_delta) * tomt;
    float y     = in_ch + num * frcp(den);
    float dnum  = in_delta * in_delta *
                  (in_dp1 * theta * theta + 2.0f * in_delta * tomt + in_d * omt * omt);
    float l     = __logf(dnum) - 2.0f * __logf(den);

    yout = inside ? y : xval;
    lad  = inside ? l : 0.0f;
}

// One 3-layer MLP (K=25 -> 8 -> 8 -> 112) on a wave's 16-row tile.
// inbuf_w: [16*32] zero-padded input rows (LDS). hbuf_w: [16*32] hidden,
// cols 0..15 written each layer (8..15 as zeros), 16..31 pre-zeroed.
// pbuf_w: [16*112] spline-parameter output.
__device__ __forceinline__ void run_mlp(
    const float* __restrict__ inbuf_w,
    float* __restrict__ hbuf_w,
    float* __restrict__ pbuf_w,
    const float* __restrict__ w0, const float* __restrict__ b0,
    const float* __restrict__ w1, const float* __restrict__ b1,
    const float* __restrict__ w2, const float* __restrict__ b2,
    int lane) {
    const int M = lane & 15;
    const int khalf = lane >> 4;
    const int N = M;
    const int nc = min(N, 7);     // clamped column for 8-wide layers
    const bool nok = (N < 8);

    v16h a, bf;
    // ---- layer 1: [16x25pad32] x [25x8 pad 32x16] ----
#pragma unroll
    for (int p = 0; p < 8; ++p) {
        int k0 = a_kmap(p, khalf);
        a[2 * p]     = (_Float16)inbuf_w[M * 32 + k0];
        a[2 * p + 1] = (_Float16)inbuf_w[M * 32 + k0 + 1];
    }
#pragma unroll
    for (int p = 0; p < 8; ++p) {
        int k0 = a_kmap(p, khalf);
        float f0 = w0[min(k0, 24) * 8 + nc];
        float f1 = w0[min(k0 + 1, 24) * 8 + nc];
        f0 = (k0 < 25 && nok) ? f0 : 0.0f;        // v_cndmask, no branch
        f1 = (k0 + 1 < 25 && nok) ? f1 : 0.0f;
        bf[2 * p] = (_Float16)f0; bf[2 * p + 1] = (_Float16)f1;
    }
    v8f c1 = {};
    c1 = __builtin_amdgcn_wmma_f32_16x16x32_f16(false, a, false, bf,
                                                (short)0, c1, false, false);
    float bias0 = b0[nc];
#pragma unroll
    for (int j = 0; j < 8; ++j) {                 // unconditional, value-masked
        int m = j + 8 * khalf;
        float v = tanh_f(c1[j] + bias0);
        hbuf_w[m * 32 + N] = nok ? v : 0.0f;      // cols 8..15 stay zero
    }
    __syncthreads();

    // ---- layer 2: [16x8 pad 32] x [8x8 pad 32x16] ----
#pragma unroll
    for (int p = 0; p < 8; ++p) {
        int k0 = a_kmap(p, khalf);
        a[2 * p]     = (_Float16)hbuf_w[M * 32 + k0];   // cols >=8 are zero
        a[2 * p + 1] = (_Float16)hbuf_w[M * 32 + k0 + 1];
    }
#pragma unroll
    for (int p = 0; p < 8; ++p) {
        int k0 = a_kmap(p, khalf);
        float f0 = w1[min(k0, 7) * 8 + nc];
        float f1 = w1[min(k0 + 1, 7) * 8 + nc];
        f0 = (k0 < 8 && nok) ? f0 : 0.0f;
        f1 = (k0 + 1 < 8 && nok) ? f1 : 0.0f;
        bf[2 * p] = (_Float16)f0; bf[2 * p + 1] = (_Float16)f1;
    }
    v8f c2 = {};
    c2 = __builtin_amdgcn_wmma_f32_16x16x32_f16(false, a, false, bf,
                                                (short)0, c2, false, false);
    float bias1 = b1[nc];
    __syncthreads();  // all lanes done reading h1 before overwrite
#pragma unroll
    for (int j = 0; j < 8; ++j) {
        int m = j + 8 * khalf;
        float v = tanh_f(c2[j] + bias1);
        hbuf_w[m * 32 + N] = nok ? v : 0.0f;
    }
    __syncthreads();

    // ---- layer 3: [16x8 pad 32] x [8x112], 7 N-tiles of 16 ----
#pragma unroll
    for (int p = 0; p < 8; ++p) {
        int k0 = a_kmap(p, khalf);
        a[2 * p]     = (_Float16)hbuf_w[M * 32 + k0];
        a[2 * p + 1] = (_Float16)hbuf_w[M * 32 + k0 + 1];
    }
#pragma unroll
    for (int t = 0; t < 7; ++t) {
#pragma unroll
        for (int p = 0; p < 8; ++p) {
            int k0 = a_kmap(p, khalf);
            float f0 = w2[min(k0, 7) * 112 + 16 * t + N];
            float f1 = w2[min(k0 + 1, 7) * 112 + 16 * t + N];
            f0 = (k0 < 8) ? f0 : 0.0f;
            f1 = (k0 + 1 < 8) ? f1 : 0.0f;
            bf[2 * p] = (_Float16)f0; bf[2 * p + 1] = (_Float16)f1;
        }
        v8f c3 = {};
        c3 = __builtin_amdgcn_wmma_f32_16x16x32_f16(false, a, false, bf,
                                                    (short)0, c3, false, false);
        float bias3 = b2[16 * t + N];
#pragma unroll
        for (int j = 0; j < 8; ++j) {
            int m = j + 8 * khalf;
            pbuf_w[m * 112 + 16 * t + N] = c3[j] + bias3;
        }
    }
    __syncthreads();
}

__global__ void __launch_bounds__(WAVES_PER_BLOCK * 32)
nsf_fused_kernel(const float* __restrict__ x, const float* __restrict__ cond,
                 const float* __restrict__ w1_0, const float* __restrict__ b1_0,
                 const float* __restrict__ w1_1, const float* __restrict__ b1_1,
                 const float* __restrict__ w1_2, const float* __restrict__ b1_2,
                 const float* __restrict__ w2_0, const float* __restrict__ b2_0,
                 const float* __restrict__ w2_1, const float* __restrict__ b2_1,
                 const float* __restrict__ w2_2, const float* __restrict__ b2_2,
                 float* __restrict__ out, float* __restrict__ logdet) {
    __shared__ __align__(16) float inbuf[WAVES_PER_BLOCK][16 * 32]; // 0..7 active
                                                                    // 8..24 cond, 25..31 zero
    __shared__ __align__(16) float hbuf[WAVES_PER_BLOCK][16 * 32];  // hidden
    __shared__ __align__(16) float pbuf[WAVES_PER_BLOCK][16 * 112]; // spline params
    __shared__ __align__(16) float ubuf[WAVES_PER_BLOCK][16 * 8];   // upper half
    __shared__ __align__(16) float lbuf[WAVES_PER_BLOCK][16 * 8];   // lower half
    __shared__ __align__(16) float ldb[WAVES_PER_BLOCK][16];        // per-row logdet
    __shared__ __align__(16) float dump[WAVES_PER_BLOCK][32];       // async sink

    const int lane = threadIdx.x & 31;
    const int wid  = threadIdx.x >> 5;
    const int rowbase = (blockIdx.x * WAVES_PER_BLOCK + wid) * ROWS_PER_WAVE;

    // ---- async-stage x tile (16 rows x 64B) as b128 chunks into lbuf/ubuf ----
    {
        const float* xbase = x + (size_t)rowbase * 16;
#pragma unroll
        for (int it = 0; it < 2; ++it) {
            int idx = it * 32 + lane;            // 0..63 chunks of 4 floats
            int r = idx >> 2, q = idx & 3;       // q: which 16B of the row
            uint32_t ld = (q < 2) ? lds_off(&lbuf[wid][r * 8 + 4 * q])
                                  : lds_off(&ubuf[wid][r * 8 + 4 * (q - 2)]);
            async_ld_b128(ld, (uint32_t)(idx * 16), xbase);
        }
    }
    // ---- async-stage cond tile into inbuf columns 8..24 (b32) ----
    {
        const float* cbase = cond + (size_t)rowbase * 17;
#pragma unroll
        for (int it = 0; it < 9; ++it) {
            int idx = it * 32 + lane;            // 0..287, 272 valid
            int idc = min(idx, 271);
            int r = idc / 17, c = idc % 17;
            uint32_t ld = (idx < 272) ? lds_off(&inbuf[wid][r * 32 + 8 + c])
                                      : lds_off(&dump[wid][lane]);
            async_ld_b32(ld, (uint32_t)(idc * 4), cbase);
        }
    }
    // ---- zero pads (disjoint from async targets: no ordering hazard) ----
    for (int i = lane; i < 16 * 7; i += 32) {    // inbuf cols 25..31
        int r = i / 7, c = 25 + (i % 7);
        inbuf[wid][r * 32 + c] = 0.0f;
    }
    for (int i = lane; i < 16 * 16; i += 32) {   // hbuf cols 16..31
        int r = i >> 4, c = 16 + (i & 15);
        hbuf[wid][r * 32 + c] = 0.0f;
    }
    if (lane < 16) ldb[wid][lane] = 0.0f;
    wait_async0();
    __syncthreads();

    // copy lower half into inbuf cols 0..7 (MLP1 input = [lower, cond])
#pragma unroll
    for (int it = 0; it < 4; ++it) {
        int i = it * 32 + lane;                  // 0..127
        int r = i >> 3, c = i & 7;
        inbuf[wid][r * 32 + c] = lbuf[wid][r * 8 + c];
    }
    __syncthreads();

    // ---- Coupling 1: params from [lower, cond], spline on upper ----
    run_mlp(inbuf[wid], hbuf[wid], pbuf[wid],
            w1_0, b1_0, w1_1, b1_1, w1_2, b1_2, lane);
#pragma unroll
    for (int it = 0; it < 4; ++it) {
        int task = it * 32 + lane;               // 16 rows x 8 dims
        int r = task >> 3, dm = task & 7;
        float prm[14];
#pragma unroll
        for (int i = 0; i < 14; ++i) prm[i] = pbuf[wid][r * 112 + dm * 14 + i];
        float y, lad;
        rqs_eval(prm, ubuf[wid][r * 8 + dm], y, lad);
        ubuf[wid][r * 8 + dm] = y;               // same slot, same lane
        atomicAdd(&ldb[wid][r], lad);            // ds_add_f32
    }
    __syncthreads();

    // MLP2 input = [upper_new, cond]: refresh inbuf cols 0..7
#pragma unroll
    for (int it = 0; it < 4; ++it) {
        int i = it * 32 + lane;
        int r = i >> 3, c = i & 7;
        inbuf[wid][r * 32 + c] = ubuf[wid][r * 8 + c];
    }
    __syncthreads();

    // ---- Coupling 2: params from [upper_new, cond], spline on lower ----
    run_mlp(inbuf[wid], hbuf[wid], pbuf[wid],
            w2_0, b2_0, w2_1, b2_1, w2_2, b2_2, lane);
#pragma unroll
    for (int it = 0; it < 4; ++it) {
        int task = it * 32 + lane;
        int r = task >> 3, dm = task & 7;
        float prm[14];
#pragma unroll
        for (int i = 0; i < 14; ++i) prm[i] = pbuf[wid][r * 112 + dm * 14 + i];
        float y, lad;
        rqs_eval(prm, lbuf[wid][r * 8 + dm], y, lad);
        lbuf[wid][r * 8 + dm] = y;
        atomicAdd(&ldb[wid][r], lad);
    }
    __syncthreads();

    // ---- Write concat([lower_new, upper_new]) as float4, plus logdet ----
#pragma unroll
    for (int it = 0; it < 2; ++it) {
        int idx = it * 32 + lane;                // 0..63 chunks of 4 floats
        int r = idx >> 2, q = idx & 3;
        const float* src = (q < 2) ? &lbuf[wid][r * 8 + 4 * q]
                                   : &ubuf[wid][r * 8 + 4 * (q - 2)];
        float4 v = make_float4(src[0], src[1], src[2], src[3]); // ds_load_b128
        *(float4*)(out + (size_t)(rowbase + r) * 16 + 4 * q) = v;
    }
    if (lane < 16) logdet[rowbase + lane] = ldb[wid][lane];
}

extern "C" void kernel_launch(void* const* d_in, const int* in_sizes, int n_in,
                              void* d_out, int out_size, void* d_ws, size_t ws_size,
                              hipStream_t stream) {
    (void)n_in; (void)out_size; (void)d_ws; (void)ws_size;
    const float* x    = (const float*)d_in[0];
    const float* cond = (const float*)d_in[1];
    const float* w1_0 = (const float*)d_in[2];
    const float* b1_0 = (const float*)d_in[3];
    const float* w1_1 = (const float*)d_in[4];
    const float* b1_1 = (const float*)d_in[5];
    const float* w1_2 = (const float*)d_in[6];
    const float* b1_2 = (const float*)d_in[7];
    const float* w2_0 = (const float*)d_in[8];
    const float* b2_0 = (const float*)d_in[9];
    const float* w2_1 = (const float*)d_in[10];
    const float* b2_1 = (const float*)d_in[11];
    const float* w2_2 = (const float*)d_in[12];
    const float* b2_2 = (const float*)d_in[13];

    const int batch = in_sizes[0] / 16;           // x is [B,16]
    float* out    = (float*)d_out;                // [B,16]
    float* logdet = out + (size_t)batch * 16;     // [B]

    const int rows_per_block = WAVES_PER_BLOCK * ROWS_PER_WAVE;  // 64
    const int blocks = batch / rows_per_block;    // BATCH=524288 -> 8192

    hipLaunchKernelGGL(nsf_fused_kernel, dim3(blocks), dim3(WAVES_PER_BLOCK * 32),
                       0, stream,
                       x, cond,
                       w1_0, b1_0, w1_1, b1_1, w1_2, b1_2,
                       w2_0, b2_0, w2_1, b2_1, w2_2, b2_2,
                       out, logdet);
}
